// CrossAttention_61804579389767
// MI455X (gfx1250) — matrix operand order repaired and checked by reading, not compiled, not run on hardware
//
#include <hip/hip_runtime.h>
#include <hip/hip_bf16.h>

// CDNA5 (gfx1250) wave32 WMMA implementation of cross-attention.
// Matmuls: v_wmma_f32_16x16x32_bf16 (bf16 in, fp32 accumulate).
// LDS staging: GLOBAL_LOAD_ASYNC_TO_LDS_B128 (ASYNCcnt) when available,
// double-buffered in the GEMM K-loop.

typedef __attribute__((ext_vector_type(16))) __bf16 v16bf;
typedef __attribute__((ext_vector_type(8)))  float  v8f;
typedef __attribute__((ext_vector_type(4)))  int    v4i;

#define WMMA_BF16(a, b, c) \
  __builtin_amdgcn_wmma_f32_16x16x32_bf16(false, (a), false, (b), (short)0, (c), false, false)

#if __has_builtin(__builtin_amdgcn_global_load_async_to_lds_b128) && \
    __has_builtin(__builtin_amdgcn_s_wait_asynccnt)
#define USE_ASYNC_LDS 1
#else
#define USE_ASYNC_LDS 0
#endif

// Copy 16 bytes (8 bf16) global -> LDS. Async path uses the CDNA5
// async-to-LDS unit (tracked by ASYNCcnt); fallback is a plain vector copy.
// Toolchain signature: (v4i AS1*, v4i AS3*, imm offset, imm cpol).
// AS3 pointers are 32-bit LDS offsets; the generic LDS address carries the
// offset in its low 32 bits (ISA 10.2 aperture rules), so integer casts are
// the portable way to build both qualified pointers.
static __device__ __forceinline__ void stage16B(const __bf16* g, __bf16* l) {
#if USE_ASYNC_LDS
  __builtin_amdgcn_global_load_async_to_lds_b128(
      (__attribute__((address_space(1))) v4i*)(uintptr_t)(const void*)g,
      (__attribute__((address_space(3))) v4i*)(uint32_t)(uintptr_t)(void*)l,
      0, 0);
#else
  *(uint4*)l = *(const uint4*)g;
#endif
}

static __device__ __forceinline__ void wait_async_copies() {
#if USE_ASYNC_LDS
  __builtin_amdgcn_s_wait_asynccnt(0);
#endif
}

static __device__ __forceinline__ v8f zerov8() {
  v8f z = {0.f, 0.f, 0.f, 0.f, 0.f, 0.f, 0.f, 0.f};
  return z;
}

// A-matrix fragment (16x32 bf16, row-major source, stride ld elements).
// Layout: lane L holds row r = L%16; lanes 0-15: K in {0..15},
// lanes 16-31: K in {16..31}; elements packed 2/VGPR.
static __device__ __forceinline__ v16bf load_frag_a(const __bf16* base, int ld) {
  const int lane = threadIdx.x & 31;
  const int r = lane & 15, hi = lane >> 4;
  const __bf16* p = base + r * ld + 16 * hi;
  v16bf a;
#pragma unroll
  for (int e = 0; e < 8; ++e) a[e] = p[e];
#pragma unroll
  for (int e = 0; e < 8; ++e) a[8 + e] = p[8 + e];
  return a;
}

// B-matrix fragment (32x16) from B-transposed storage: row n holds K
// contiguously (i.e. weight W[n][k]). Lane L holds column n = L%16,
// K = e + 16*(L>>4) -> one contiguous 32-byte chunk per lane.
static __device__ __forceinline__ v16bf load_frag_bt(const __bf16* base, int ld) {
  const int lane = threadIdx.x & 31;
  const int n = lane & 15, hi = lane >> 4;
  const __bf16* p = base + n * ld + 16 * hi;
  v16bf b;
#pragma unroll
  for (int e = 0; e < 16; ++e) b[e] = p[e];
  return b;
}

// B-matrix fragment from natural [K x N] storage (stride ld).
static __device__ __forceinline__ v16bf load_frag_bn(const __bf16* base, int ld) {
  const int lane = threadIdx.x & 31;
  const int n = lane & 15, hi = lane >> 4;
  v16bf b;
#pragma unroll
  for (int e = 0; e < 16; ++e) b[e] = base[(e + 16 * hi) * ld + n];
  return b;
}

__global__ void cvt_f32_bf16(const float* __restrict__ s, __bf16* __restrict__ d, long n) {
  long i = (long)blockIdx.x * blockDim.x + threadIdx.x;
  long stride = (long)gridDim.x * blockDim.x;
  for (; i < n; i += stride) d[i] = (__bf16)s[i];
}

// Tiled GEMM: D[M,N] = A[M,K] * W[N,K]^T + bias[N].
// 64x64 tile / workgroup, 4 waves each computing a 32x32 sub-tile as 2x2
// 16x16 WMMA fragments; K staged as double-buffered 32-wide LDS slabs fed
// by the async-to-LDS unit. M,N,K multiples of 64.
#define BK  32
#define LDT 40  // padded LDS row stride (bf16 elems) to avoid bank conflicts

template <bool OUT_F32>
__global__ __launch_bounds__(128) void gemm_bf16_nt(
    const __bf16* __restrict__ A, const __bf16* __restrict__ W,
    const float* __restrict__ bias, float* __restrict__ Of,
    __bf16* __restrict__ Ob, int M, int N, int K) {
  __shared__ __bf16 As[2][64 * LDT];
  __shared__ __bf16 Ws[2][64 * LDT];
  const int t = threadIdx.x;
  const int lane = t & 31, w = t >> 5;
  const int m0 = blockIdx.y * 64, n0 = blockIdx.x * 64;
  const int wr = (w >> 1) * 32, wc = (w & 1) * 32;

  v8f acc00 = zerov8(), acc01 = zerov8(), acc10 = zerov8(), acc11 = zerov8();

  const int lr = t >> 1, lc = (t & 1) * 16;  // cooperative loader coords

  // issue one 32-wide K slab (A + W tiles) into LDS buffer `buf`
  auto stage_slab = [&](int buf, int k0) {
    const __bf16* ag = A + (size_t)(m0 + lr) * K + k0 + lc;
    const __bf16* wg = W + (size_t)(n0 + lr) * K + k0 + lc;
    __bf16* as = &As[buf][lr * LDT + lc];
    __bf16* ws = &Ws[buf][lr * LDT + lc];
    stage16B(ag, as);
    stage16B(ag + 8, as + 8);
    stage16B(wg, ws);
    stage16B(wg + 8, ws + 8);
  };

  stage_slab(0, 0);
  for (int k0 = 0; k0 < K; k0 += BK) {
    const int cur = (k0 / BK) & 1;
    wait_async_copies();   // my slab `cur` copies are in LDS
    __syncthreads();       // everyone's are, and prior reads of `cur^1` done
    if (k0 + BK < K) stage_slab(cur ^ 1, k0 + BK);

    v16bf a0 = load_frag_a(&As[cur][(wr + 0) * LDT], LDT);
    v16bf a1 = load_frag_a(&As[cur][(wr + 16) * LDT], LDT);
    v16bf b0 = load_frag_bt(&Ws[cur][(wc + 0) * LDT], LDT);
    v16bf b1 = load_frag_bt(&Ws[cur][(wc + 16) * LDT], LDT);
    acc00 = WMMA_BF16(a0, b0, acc00);
    acc01 = WMMA_BF16(a0, b1, acc01);
    acc10 = WMMA_BF16(a1, b0, acc10);
    acc11 = WMMA_BF16(a1, b1, acc11);
  }

  const int cn = lane & 15, hi = lane >> 4;
  v8f* accs[2][2] = {{&acc00, &acc01}, {&acc10, &acc11}};
#pragma unroll
  for (int i = 0; i < 2; ++i)
#pragma unroll
    for (int j = 0; j < 2; ++j)
#pragma unroll
      for (int v = 0; v < 8; ++v) {
        int row = m0 + wr + i * 16 + v + 8 * hi;
        int col = n0 + wc + j * 16 + cn;
        float val = (*accs[i][j])[v] + bias[col];
        if (OUT_F32) Of[(size_t)row * N + col] = val;
        else         Ob[(size_t)row * N + col] = (__bf16)val;
      }
}

// Fused attention. Workgroup = (batch b, 32-query tile), loops all 16 heads.
// Per head: S = Q K^T (WMMA) -> masked softmax (all 128 lanes, shfl-reduced)
// -> att_mean accumulation (exclusive ownership, no atomics) -> y = P V
// (WMMA) -> Y (bf16).
// Dynamic LDS: S 32x1024 f32 (128K) + P 32x1024 bf16 (64K) + tile 32x64 (4K).
__global__ __launch_bounds__(128) void attn_fused(
    const __bf16* __restrict__ Qf, const __bf16* __restrict__ Kf,
    const __bf16* __restrict__ Vf, const unsigned char* __restrict__ mask,
    float* __restrict__ att_mean, __bf16* __restrict__ Yb,
    int T, int TE, int C) {
  extern __shared__ __align__(16) char smem[];
  float*  S  = (float*)smem;                                    // [32][1024]
  __bf16* P  = (__bf16*)(smem + 32 * 1024 * 4);                 // [32][1024]
  __bf16* QV = (__bf16*)(smem + 32 * 1024 * 4 + 32 * 1024 * 2); // [32][64]

  const int t = threadIdx.x;
  const int lane = t & 31, w = t >> 5;
  const int m0 = blockIdx.x * 32;
  const int b  = blockIdx.y;
  const int cn = lane & 15, hi = lane >> 4;
  const float scale = 0.125f;            // 1/sqrt(64)
  const float MINV = -3.4028235e38f;     // matches jnp.finfo(f32).min

  // stage a [32 x 64] bf16 tile (rows row0..row0+31 of src, cols col0..+63)
  auto stage_tile = [&](const __bf16* src, size_t row0, int col0) {
    int r = t >> 2, c0 = (t & 3) * 16;
    const __bf16* g = src + (row0 + r) * C + col0 + c0;
    __bf16* l = QV + r * 64 + c0;
    stage16B(g, l);
    stage16B(g + 8, l + 8);
  };

  for (int h = 0; h < 16; ++h) {
    __syncthreads();
    // 1) Q tile [32 x 64]
    stage_tile(Qf, (size_t)(b * T + m0), h * 64);
    wait_async_copies();
    __syncthreads();

    // 2) S = Q K^T ; wave w owns 256 score columns
    for (int nt = 0; nt < 16; ++nt) {
      int n0 = w * 256 + nt * 16;
      v8f acc0 = zerov8(), acc1 = zerov8();
#pragma unroll
      for (int ks = 0; ks < 2; ++ks) {
        int k0 = ks * 32;
        v16bf a0 = load_frag_a(QV + 0 * 64 + k0, 64);
        v16bf a1 = load_frag_a(QV + 16 * 64 + k0, 64);
        // K rows are K-dim contiguous in memory -> B^T style fragment
        v16bf bk = load_frag_bt(Kf + (size_t)(b * TE + n0) * C + h * 64 + k0, C);
        acc0 = WMMA_BF16(a0, bk, acc0);
        acc1 = WMMA_BF16(a1, bk, acc1);
      }
#pragma unroll
      for (int v = 0; v < 8; ++v) {
        S[(0 + v + 8 * hi) * 1024 + n0 + cn]  = acc0[v];
        S[(16 + v + 8 * hi) * 1024 + n0 + cn] = acc1[v];
      }
    }
    __syncthreads();

    // 3) masked softmax + att_mean: 4 lanes per row, 256 columns each;
    //    row max/sum reduced with wave32 shfl_xor (4-lane groups are
    //    contiguous lanes within one wave).
    {
      const int r  = t >> 2;            // row 0..31
      const int qq = t & 3;             // segment 0..3
      const int seg = TE >> 2;
      const int s0 = qq * seg, s1 = s0 + seg;
      const unsigned char* mrow = mask + (size_t)(b * T + m0 + r) * TE;
      float* srow = S + r * 1024;

      float mx = MINV;
      for (int s = s0; s < s1; ++s) {
        float v = mrow[s] ? MINV : srow[s] * scale;
        srow[s] = v;
        mx = fmaxf(mx, v);
      }
      mx = fmaxf(mx, __shfl_xor(mx, 1));
      mx = fmaxf(mx, __shfl_xor(mx, 2));

      float sum = 0.f;
      for (int s = s0; s < s1; ++s) {
        float e = __expf(srow[s] - mx);
        srow[s] = e;
        sum += e;
      }
      sum += __shfl_xor(sum, 1);
      sum += __shfl_xor(sum, 2);

      const float inv = 1.0f / sum;
      float* am = att_mean + (size_t)(b * T + m0 + r) * TE;
      for (int s = s0; s < s1; ++s) {
        float p = mrow[s] ? 0.f : srow[s] * inv;
        P[r * 1024 + s] = (__bf16)p;
        float contrib = p * 0.0625f;  // /H
        am[s] = (h == 0) ? contrib : (am[s] + contrib);
      }
    }
    __syncthreads();

    // 4) y = P @ V ; wave w owns 16 output columns, K loop over TE
    v8f y0 = zerov8(), y1 = zerov8();
    for (int ks = 0; ks < 32; ++ks) {
      int s0 = ks * 32;
      __syncthreads();  // previous tile reads complete
      stage_tile(Vf, (size_t)(b * TE + s0), h * 64);
      wait_async_copies();
      __syncthreads();
      v16bf a0 = load_frag_a(P + 0 * 1024 + s0, 1024);
      v16bf a1 = load_frag_a(P + 16 * 1024 + s0, 1024);
      v16bf bv = load_frag_bn(QV + w * 16, 64);
      y0 = WMMA_BF16(a0, bv, y0);
      y1 = WMMA_BF16(a1, bv, y1);
    }
#pragma unroll
    for (int v = 0; v < 8; ++v) {
      int col = h * 64 + w * 16 + cn;
      Yb[(size_t)(b * T + m0 + 0 + v + 8 * hi) * C + col]  = (__bf16)y0[v];
      Yb[(size_t)(b * T + m0 + 16 + v + 8 * hi) * C + col] = (__bf16)y1[v];
    }
  }
}

extern "C" void kernel_launch(void* const* d_in, const int* in_sizes, int n_in,
                              void* d_out, int out_size, void* d_ws, size_t ws_size,
                              hipStream_t stream) {
  (void)in_sizes; (void)n_in; (void)out_size; (void)ws_size;
  const int B = 4, T = 2048, TE = 1024, C = 1024;

  const float* x   = (const float*)d_in[0];
  const float* enc = (const float*)d_in[1];
  const unsigned char* mask = (const unsigned char*)d_in[2];  // numpy bool (1B)
  const float* Wq = (const float*)d_in[3];
  const float* bq = (const float*)d_in[4];
  const float* Wk = (const float*)d_in[5];
  const float* bk = (const float*)d_in[6];
  const float* Wv = (const float*)d_in[7];
  const float* bv = (const float*)d_in[8];
  const float* Wp = (const float*)d_in[9];
  const float* bp = (const float*)d_in[10];

  // workspace carve-out (bf16 intermediates), 256B aligned
  char* ws = (char*)d_ws;
  size_t off = 0;
  auto carve = [&](size_t bytes) {
    char* p = ws + off;
    off = (off + bytes + 255) & ~(size_t)255;
    return p;
  };
  __bf16* xb  = (__bf16*)carve((size_t)B * T * C * 2);
  __bf16* eb  = (__bf16*)carve((size_t)B * TE * C * 2);
  __bf16* wqb = (__bf16*)carve((size_t)C * C * 2);
  __bf16* wkb = (__bf16*)carve((size_t)C * C * 2);
  __bf16* wvb = (__bf16*)carve((size_t)C * C * 2);
  __bf16* wpb = (__bf16*)carve((size_t)C * C * 2);
  __bf16* Qb  = (__bf16*)carve((size_t)B * T * C * 2);
  __bf16* Kb  = (__bf16*)carve((size_t)B * TE * C * 2);
  __bf16* Vb  = (__bf16*)carve((size_t)B * TE * C * 2);
  __bf16* Yb  = (__bf16*)carve((size_t)B * T * C * 2);

  float* out_y   = (float*)d_out;
  float* out_att = out_y + (size_t)B * T * C;

  dim3 blk(128);
  // f32 -> bf16 conversions
  cvt_f32_bf16<<<2048, 256, 0, stream>>>(x, xb, (long)B * T * C);
  cvt_f32_bf16<<<1024, 256, 0, stream>>>(enc, eb, (long)B * TE * C);
  cvt_f32_bf16<<<512, 256, 0, stream>>>(Wq, wqb, (long)C * C);
  cvt_f32_bf16<<<512, 256, 0, stream>>>(Wk, wkb, (long)C * C);
  cvt_f32_bf16<<<512, 256, 0, stream>>>(Wv, wvb, (long)C * C);
  cvt_f32_bf16<<<512, 256, 0, stream>>>(Wp, wpb, (long)C * C);

  // Q / K / V projections (bf16 out)
  gemm_bf16_nt<false><<<dim3(C / 64, (B * T) / 64), blk, 0, stream>>>(
      xb, wqb, bq, nullptr, Qb, B * T, C, C);
  gemm_bf16_nt<false><<<dim3(C / 64, (B * TE) / 64), blk, 0, stream>>>(
      eb, wkb, bk, nullptr, Kb, B * TE, C, C);
  gemm_bf16_nt<false><<<dim3(C / 64, (B * TE) / 64), blk, 0, stream>>>(
      eb, wvb, bv, nullptr, Vb, B * TE, C, C);

  // fused attention (writes att_mean directly, Y in bf16)
  size_t shmem = (size_t)32 * 1024 * 4 + 32 * 1024 * 2 + 32 * 64 * 2;  // 200704 B
  attn_fused<<<dim3(T / 32, B), blk, shmem, stream>>>(
      Qb, Kb, Vb, mask, out_att, Yb, T, TE, C);

  // output projection (fp32 out)
  gemm_bf16_nt<true><<<dim3(C / 64, (B * T) / 64), blk, 0, stream>>>(
      Yb, wpb, bp, out_y, nullptr, B * T, C, C);
}
